// FixedFMoE_25486335934680
// MI455X (gfx1250) — matrix operand order repaired and checked by reading, not compiled, not run on hardware
//
#include <hip/hip_runtime.h>
#include <math.h>

#define NUM_EXPERT 32
#define D_MODEL    1024
#define D_HIDDEN   4096
#define TOPK       2
#define TOKENS     4096
#define CAP        512
#define A_TOTAL    (TOKENS * TOPK)   // 8192 assignments

typedef __attribute__((ext_vector_type(16))) __bf16 v16bf;
typedef __attribute__((ext_vector_type(8)))  __bf16 v8bf;
typedef __attribute__((ext_vector_type(4)))  __bf16 v4bf;
typedef __attribute__((ext_vector_type(8)))  float  v8f;

// Assemble a 16-element bf16 fragment from two aligned 16-byte LDS chunks.
__device__ __forceinline__ v16bf ld_frag(const __bf16* p0, const __bf16* p1) {
  v8bf lo = *(const v8bf*)p0;
  v8bf hi = *(const v8bf*)p1;
  v16bf r;
#pragma unroll
  for (int i = 0; i < 8; ++i) { r[i] = lo[i]; r[i + 8] = hi[i]; }
  return r;
}

// Convert 4 floats -> packed v4bf and store as one 8-byte LDS store.
__device__ __forceinline__ void cvt_store4(__bf16* dst, float a, float b,
                                           float c, float d) {
  v4bf p; p[0] = (__bf16)a; p[1] = (__bf16)b; p[2] = (__bf16)c; p[3] = (__bf16)d;
  *(v4bf*)dst = p;
}

// Load a 64(K) x 128(N) fp32 weight slab into registers: 8 b128 loads, all
// outstanding together (t[] forces distinct register quads).
template <int N>
__device__ __forceinline__ void ld_tile(const float* base, float4 t[2][4]) {
#pragma unroll
  for (int i = 0; i < 2; ++i)
#pragma unroll
    for (int r = 0; r < 4; ++r)
      t[i][r] = *(const float4*)(base + (size_t)(i * 32 + r) * N);
}

// Transpose-convert slab into swizzled nt[n][k] bf16 tile: 8 ds_store_b64.
__device__ __forceinline__ void st_tile(__bf16* nt, const int* ntaddr,
                                        float4 t[2][4]) {
#pragma unroll
  for (int i = 0; i < 2; ++i)
#pragma unroll
    for (int j = 0; j < 4; ++j)
      cvt_store4(nt + ntaddr[i * 4 + j],
                 ((const float*)&t[i][0])[j], ((const float*)&t[i][1])[j],
                 ((const float*)&t[i][2])[j], ((const float*)&t[i][3])[j]);
}

// ---------------------------------------------------------------- init ----
__global__ void moe_init_kernel(int* counts) {
  if (threadIdx.x < NUM_EXPERT) counts[threadIdx.x] = 0;
}

// ------------------------------------------------------- gate + dispatch ----
__global__ __launch_bounds__(256)
void moe_gate_dispatch_kernel(const float* __restrict__ x,
                              const float* __restrict__ Wg,
                              const float* __restrict__ bg,
                              int* counts, int* valid, float* gateW,
                              int* slotToken, int* slotAssign) {
  int wave = (blockIdx.x * blockDim.x + threadIdx.x) >> 5;
  int lane = threadIdx.x & 31;
  if (wave >= TOKENS) return;
  const float* xr = x + (size_t)wave * D_MODEL;

  float acc = 0.f;
  for (int i = 0; i < D_MODEL; ++i)
    acc = fmaf(xr[i], Wg[i * NUM_EXPERT + lane], acc);
  acc += bg[lane];

  float m1 = acc; int i1 = lane;
#pragma unroll
  for (int off = 16; off >= 1; off >>= 1) {
    float ov = __shfl_xor(m1, off, 32);
    int   oi = __shfl_xor(i1, off, 32);
    if (ov > m1 || (ov == m1 && oi < i1)) { m1 = ov; i1 = oi; }
  }
  float v2 = (lane == i1) ? -__builtin_inff() : acc;
  float m2 = v2; int i2 = lane;
#pragma unroll
  for (int off = 16; off >= 1; off >>= 1) {
    float ov = __shfl_xor(m2, off, 32);
    int   oi = __shfl_xor(i2, off, 32);
    if (ov > m2 || (ov == m2 && oi < i2)) { m2 = ov; i2 = oi; }
  }
  float e2 = __expf(m2 - m1);
  float g0 = 1.f / (1.f + e2);
  float g1 = e2 / (1.f + e2);

  if (lane == 0) {
    int t = wave;
    int   es[2] = { i1, i2 };
    float gs[2] = { g0, g1 };
#pragma unroll
    for (int k = 0; k < TOPK; ++k) {
      int a = t * TOPK + k;
      int e = es[k];
      gateW[a] = gs[k];
      int pos = atomicAdd(&counts[e], 1);
      if (pos < CAP) {
        int slot = e * CAP + pos;
        slotToken[slot]  = t;
        slotAssign[slot] = a;
        valid[a] = 1;
      } else {
        valid[a] = 0;
      }
    }
  }
}

// ------------------------------------------------------------- expert FFN ----
// LDS: xa 16x1024 bf16 (32KB) | Hb 16x2048 bf16 (64KB) | nt 128x64 bf16 (16KB)
#define SM_XA 0
#define SM_HB 32768
#define SM_NT (32768 + 65536)
#define SM_IX (32768 + 65536 + 16384)
#define SMEM_BYTES (32768 + 65536 + 16384 + 128)

__global__ __launch_bounds__(256)
void moe_ffn_kernel(const float* __restrict__ x,
                    const float* __restrict__ W1, const float* __restrict__ b1,
                    const float* __restrict__ W2, const float* __restrict__ b2,
                    const int* __restrict__ counts,
                    const int* __restrict__ slotToken,
                    const int* __restrict__ slotAssign,
                    float* __restrict__ yA) {
  extern __shared__ char smem[];
  __bf16* xa = (__bf16*)(smem + SM_XA);   // swizzled [m][k], row 1024
  __bf16* Hb = (__bf16*)(smem + SM_HB);   // swizzled [m][k], row 2048
  __bf16* nt = (__bf16*)(smem + SM_NT);   // swizzled [n][k], row 64
  int* rowTok = (int*)(smem + SM_IX);
  int* rowAsn = rowTok + 16;

  int e    = blockIdx.x >> 5;
  int tile = blockIdx.x & 31;
  int cnt = counts[e]; if (cnt > CAP) cnt = CAP;
  int row0 = tile * 16;
  if (row0 >= cnt) return;
  int rows = cnt - row0; if (rows > 16) rows = 16;

  int tid  = threadIdx.x;
  int lane = tid & 31;
  int wv   = tid >> 5;
  int hf   = lane >> 4;
  int mrow = lane & 15;
  int n4   = tid & 31;   // float4 column owned during staging
  int kqb  = tid >> 5;   // base K-quad owned during staging

  if (tid < 16) {
    int m = tid;
    rowTok[m] = (m < rows) ? slotToken[e * CAP + row0 + m]  : -1;
    rowAsn[m] = (m < rows) ? slotAssign[e * CAP + row0 + m] : -1;
  }
  __syncthreads();

  // Gather 16 token rows, fp32 -> bf16, XOR-swizzled chunks (^ m&7).
#pragma unroll
  for (int i = 0; i < 16; ++i) {
    int idx = i * 256 + tid;
    int m = idx >> 8, c4 = idx & 255;
    int tok = rowTok[m];
    float4 v = (tok >= 0) ? ((const float4*)(x + (size_t)tok * D_MODEL))[c4]
                          : make_float4(0.f, 0.f, 0.f, 0.f);
    int ad = m * 1024 + ((((c4 >> 1) ^ (m & 7))) << 3) + (c4 & 1) * 4;
    cvt_store4(xa + ad, v.x, v.y, v.z, v.w);
  }
  __syncthreads();

  const float* W1e = W1 + (size_t)e * D_MODEL * D_HIDDEN;
  const float* W2e = W2 + (size_t)e * D_HIDDEN * D_MODEL;
  const float* b1e = b1 + (size_t)e * D_HIDDEN;
  const float* b2e = b2 + (size_t)e * D_MODEL;

  // Precomputed swizzled nt store addresses (loop invariant).
  int ntaddr[8];
#pragma unroll
  for (int i = 0; i < 2; ++i)
#pragma unroll
    for (int j = 0; j < 4; ++j) {
      int n  = n4 * 4 + j;
      int kq = i * 8 + kqb;
      int c  = kq >> 1;
      ntaddr[i * 4 + j] = n * 64 + ((c ^ ((n >> 2) & 7)) << 3) + 4 * (kq & 1);
    }

  // Precomputed B-fragment pointers (swizzled, 16B chunks), per s of {0,1}.
  int nn = wv * 16 + mrow;
  int bx = (nn >> 2) & 7;
  const __bf16* bp[4];
#pragma unroll
  for (int s = 0; s < 2; ++s) {
    int c0 = 4 * s + hf;
    bp[s * 2 + 0] = nt + nn * 64 + (((c0)     ^ bx) << 3);
    bp[s * 2 + 1] = nt + nn * 64 + (((c0 + 2) ^ bx) << 3);
  }
  // Precomputed A-fragment swizzled element offsets (add k0 at use site).
  int axor = mrow & 7;
  int aoff[4];
#pragma unroll
  for (int s = 0; s < 2; ++s) {
    aoff[s * 2 + 0] = (((4 * s + hf)     ^ axor) << 3);
    aoff[s * 2 + 1] = (((4 * s + 2 + hf) ^ axor) << 3);
  }

  v8f yacc[8];
#pragma unroll
  for (int j = 0; j < 8; ++j)
#pragma unroll
    for (int r = 0; r < 8; ++r) yacc[j][r] = 0.f;

  for (int hh = 0; hh < 2; ++hh) {
    // ---- phase 1: Hb = gelu(X @ W1[:, hh*2048 .. +2048) + b1) ----
    for (int nc = 0; nc < 16; ++nc) {
      int n0 = hh * 2048 + nc * 128;
      const float* wb1 = W1e + (size_t)(4 * kqb) * D_HIDDEN + n0 + n4 * 4;
      float4 t[2][4];
      ld_tile<D_HIDDEN>(wb1, t);             // slab ks=0 in flight
      v8f c;
#pragma unroll
      for (int r = 0; r < 8; ++r) c[r] = 0.f;

      for (int ks = 0; ks < 16; ++ks) {
        int k0 = ks * 64;
        __syncthreads();                     // prior readers of nt done
        st_tile(nt, ntaddr, t);
        __syncthreads();
        if (ks < 15)                         // next slab overlaps the wmmas
          ld_tile<D_HIDDEN>(wb1 + (size_t)(k0 + 64) * D_HIDDEN, t);
#pragma unroll
        for (int s = 0; s < 2; ++s) {
          const __bf16* ar = xa + mrow * 1024 + k0;
          v16bf af  = ld_frag(ar + aoff[s * 2], ar + aoff[s * 2 + 1]);
          v16bf bfr = ld_frag(bp[s * 2], bp[s * 2 + 1]);
          c = __builtin_amdgcn_wmma_f32_16x16x32_bf16(
                  false, af, false, bfr, (short)0, c, false, false);
        }
      }
      // bias + exact GELU -> swizzled Hb (each wave owns distinct columns)
      int ncol = nc * 128 + wv * 16 + mrow;
      float bias = b1e[hh * 2048 + ncol];
      int nch = ncol >> 3, nrm = ncol & 7;
#pragma unroll
      for (int r = 0; r < 8; ++r) {
        int m = r + 8 * hf;
        float vv = c[r] + bias;
        float g  = 0.5f * vv * (1.f + erff(vv * 0.70710678118654752f));
        Hb[m * 2048 + ((nch ^ (m & 7)) << 3) + nrm] = (__bf16)g;
      }
    }
    __syncthreads();

    // ---- phase 2 partial: Y += Hb(16x2048) @ W2[hh*2048 .., :] ----
    for (int nc2 = 0; nc2 < 8; ++nc2) {
      int n0 = nc2 * 128;
      const float* wb2 =
          W2e + (size_t)(hh * 2048 + 4 * kqb) * D_MODEL + n0 + n4 * 4;
      float4 t[2][4];
      ld_tile<D_MODEL>(wb2, t);
      for (int ks = 0; ks < 32; ++ks) {
        int k0 = ks * 64;
        __syncthreads();
        st_tile(nt, ntaddr, t);
        __syncthreads();
        if (ks < 31)
          ld_tile<D_MODEL>(wb2 + (size_t)(k0 + 64) * D_MODEL, t);
#pragma unroll
        for (int s = 0; s < 2; ++s) {
          const __bf16* ar = Hb + mrow * 2048 + k0;
          v16bf af  = ld_frag(ar + aoff[s * 2], ar + aoff[s * 2 + 1]);
          v16bf bfr = ld_frag(bp[s * 2], bp[s * 2 + 1]);
          yacc[nc2] = __builtin_amdgcn_wmma_f32_16x16x32_bf16(
                          false, af, false, bfr, (short)0, yacc[nc2],
                          false, false);
        }
      }
    }
    __syncthreads();   // Hb reused next half
  }

  // store Y rows to per-assignment buffer
#pragma unroll
  for (int nc2 = 0; nc2 < 8; ++nc2) {
    int n = nc2 * 128 + wv * 16 + mrow;
    float bias = b2e[n];
#pragma unroll
    for (int r = 0; r < 8; ++r) {
      int m = r + 8 * hf;
      if (m < rows) {
        int a = rowAsn[m];
        yA[(size_t)a * D_MODEL + n] = yacc[nc2][r] + bias;
      }
    }
  }
}

// ---------------------------------------------------------------- combine ----
__global__ __launch_bounds__(256)
void moe_combine_kernel(const float* __restrict__ yA,
                        const float* __restrict__ gateW,
                        const int* __restrict__ valid,
                        float* __restrict__ out) {
  int t = blockIdx.x;
  int c4 = threadIdx.x;
  float ax = 0.f, ay = 0.f, az = 0.f, aw = 0.f;
#pragma unroll
  for (int k = 0; k < TOPK; ++k) {
    int a = t * TOPK + k;
    if (valid[a]) {
      float g = gateW[a];
      float4 v = ((const float4*)(yA + (size_t)a * D_MODEL))[c4];
      ax = fmaf(g, v.x, ax); ay = fmaf(g, v.y, ay);
      az = fmaf(g, v.z, az); aw = fmaf(g, v.w, aw);
    }
  }
  float4 o = make_float4(ax, ay, az, aw);
  ((float4*)(out + (size_t)t * D_MODEL))[c4] = o;
}

// ----------------------------------------------------------------- launch ----
extern "C" void kernel_launch(void* const* d_in, const int* in_sizes, int n_in,
                              void* d_out, int out_size, void* d_ws, size_t ws_size,
                              hipStream_t stream) {
  (void)in_sizes; (void)n_in; (void)out_size; (void)ws_size;
  const float* x  = (const float*)d_in[0];
  const float* Wg = (const float*)d_in[1];
  const float* bg = (const float*)d_in[2];
  const float* W1 = (const float*)d_in[3];
  const float* b1 = (const float*)d_in[4];
  const float* W2 = (const float*)d_in[5];
  const float* b2 = (const float*)d_in[6];
  float* out = (float*)d_out;

  char* ws = (char*)d_ws;
  int*   counts     = (int*)ws;               ws += 256;
  int*   valid      = (int*)ws;               ws += (size_t)A_TOTAL * 4;
  float* gateW      = (float*)ws;             ws += (size_t)A_TOTAL * 4;
  int*   slotToken  = (int*)ws;               ws += (size_t)NUM_EXPERT * CAP * 4;
  int*   slotAssign = (int*)ws;               ws += (size_t)NUM_EXPERT * CAP * 4;
  float* yA         = (float*)ws;             // A_TOTAL * D_MODEL floats (~33.5 MB)

  moe_init_kernel<<<1, 32, 0, stream>>>(counts);
  moe_gate_dispatch_kernel<<<TOKENS / 8, 256, 0, stream>>>(
      x, Wg, bg, counts, valid, gateW, slotToken, slotAssign);
  moe_ffn_kernel<<<NUM_EXPERT * (CAP / 16), 256, SMEM_BYTES, stream>>>(
      x, W1, b1, W2, b2, counts, slotToken, slotAssign, yA);
  moe_combine_kernel<<<TOKENS, 256, 0, stream>>>(yA, gateW, valid, out);
}
